// MultiHeadAttention_50972671869450
// MI455X (gfx1250) — matrix operand order repaired and checked by reading, not compiled, not run on hardware
//
#include <hip/hip_runtime.h>

// ---------------------------------------------------------------------------
// Problem constants (fixed by the reference)
// ---------------------------------------------------------------------------
#define BATCH 2
#define SEQ   2048
#define DM    1024
#define NH    16
#define DK    64      // head dim
#define MT    (BATCH * SEQ)   // 4096 rows for the projections

typedef __attribute__((ext_vector_type(16))) __bf16 v16bf;
typedef __attribute__((ext_vector_type(8)))  __bf16 v8bf;
typedef __attribute__((ext_vector_type(4)))  __bf16 v4bf;
typedef __attribute__((ext_vector_type(8)))  float  v8f;
typedef unsigned int u32x4 __attribute__((ext_vector_type(4)));
typedef int          i32x8 __attribute__((ext_vector_type(8)));
typedef int          i32x4 __attribute__((ext_vector_type(4)));

static __device__ __forceinline__ v16bf cat8(v8bf lo, v8bf hi) {
  return __builtin_shufflevector(lo, hi, 0,1,2,3,4,5,6,7,8,9,10,11,12,13,14,15);
}

// ---------------------------------------------------------------------------
// TDM: issue a 2D bf16 tile DMA (global -> LDS) via the Tensor Data Mover.
// D# layout per CDNA5 ISA ch.8: group0 = {count, lds_addr, global_addr, type},
// group1 = {data_size, pad cfg, tensor dims, tile dims, strides}.
// Pads +4 DWORDs (8 halves) after every 32 DWORDs (64 halves = one row) so the
// LDS image matches a [rows][72] padded array.  2D tensor -> groups 2/3 zero.
// This toolchain declares the 6-arg builtin form:
//   (u32x4 g0, i32x8 g1, i32x4 g2, i32x4 g3, i32x8 extra, i32 cpol)
// ---------------------------------------------------------------------------
static __device__ __forceinline__ void tdm_load_tile_bf16(
    const __bf16* gsrc, unsigned lds_off, unsigned rows) {
  unsigned long long ga = (unsigned long long)(const void*)gsrc;
  u32x4 g0;
  g0[0] = 1u;                                          // count=1 (valid D#)
  g0[1] = lds_off;                                     // lds_addr (bytes)
  g0[2] = (unsigned)(ga & 0xffffffffu);                // global_addr[31:0]
  g0[3] = (unsigned)((ga >> 32) & 0x01ffffffu)         // global_addr[56:32]
        | (2u << 30);                                  // type = 2 ("image")
  const unsigned td0 = DM;        // tensor_dim0 (elements per full row)
  const unsigned st0 = DM;        // tensor_dim0_stride (elements)
  const unsigned tile0 = 64u;     // tile_dim0: 64 bf16 = 128 B per row
  i32x8 g1;
  g1[0] = (int)((1u << 16)                             // data_size = 2 bytes
        | (1u << 20)                                   // pad_enable
        | (4u << 22)                                   // pad_interval: 32 DW
        | (3u << 25));                                 // pad_amount: 4 DW
  g1[1] = (int)((td0 & 0xffffu) << 16);                // tensor_dim0[15:0]
  g1[2] = (int)(((td0 >> 16) & 0xffffu)                // tensor_dim0[31:16]
        | ((rows & 0xffffu) << 16));                   // tensor_dim1[15:0]
  g1[3] = (int)(((rows >> 16) & 0xffffu)               // tensor_dim1[31:16]
        | (tile0 << 16));                              // tile_dim0
  g1[4] = (int)(rows & 0xffffu);                       // tile_dim1 (tile_dim2=0)
  g1[5] = (int)st0;                                    // dim0_stride[31:0]
  g1[6] = 0;                                           // stride hi bits
  g1[7] = 0;
  i32x4 z4 = {0, 0, 0, 0};
  i32x8 z8 = {0, 0, 0, 0, 0, 0, 0, 0};
  __builtin_amdgcn_tensor_load_to_lds(g0, g1, z4, z4, z8, 0);
}

// ---------------------------------------------------------------------------
// fp32 -> bf16 conversion (vectorized x4)
// ---------------------------------------------------------------------------
__global__ void f32_to_bf16_kernel(const float* __restrict__ in,
                                   __bf16* __restrict__ out, int n4) {
  int i = blockIdx.x * blockDim.x + threadIdx.x;
  int stride = gridDim.x * blockDim.x;
  for (; i < n4; i += stride) {
    float4 f = ((const float4*)in)[i];
    v4bf o;
    o.x = (__bf16)f.x; o.y = (__bf16)f.y; o.z = (__bf16)f.z; o.w = (__bf16)f.w;
    ((v4bf*)out)[i] = o;
  }
}

// ---------------------------------------------------------------------------
// Tiled bf16 WMMA GEMM with TDM double-buffered staging:
//   out[m,n] = sum_k A[m,k] * W[n,k] + bias[n]        (x @ W^T + b)
// Block: 256 threads = 8 waves; block tile 64(M) x 128(N); K-step 64.
// Wave tile 32x32; wave 0 drives the Tensor Data Mover: while stage i is being
// consumed, the DMAs for stage i+1 are already in flight (TENSORcnt pipelining,
// in-order completion => s_wait_tensorcnt(2) releases the current pair).
// MODE: 1 = store bf16 as [B,H,S,dk] (Q/K head layout)
//       2 = store bf16 as [B,H,dk,S] (V transposed per head)
//       3 = store f32 row-major [M,N] (final output projection)
// ---------------------------------------------------------------------------
#define TM 64
#define TN 128
#define TK 64
#define LROW (TK + 8)     // 72 halves = 144 B padded row

template <int MODE>
__launch_bounds__(256)
__global__ void gemm_bf16_wmma(const __bf16* __restrict__ A,
                               const __bf16* __restrict__ W,
                               const float*  __restrict__ bias,
                               __bf16* __restrict__ outb,
                               float*  __restrict__ outf) {
  const int K = DM;
  __shared__ __align__(16) __bf16 lA[2][TM][LROW];   // 18 KB
  __shared__ __align__(16) __bf16 lB[2][TN][LROW];   // 37 KB

  const int t    = threadIdx.x;
  const int wave = t >> 5, lane = t & 31;
  const int half = lane >> 4, l16 = lane & 15;
  const int waveM = wave >> 2, waveN = wave & 3;
  const int m0 = blockIdx.y * TM, n0 = blockIdx.x * TN;

  v8f acc[2][2] = {};

  // Prime the pipeline: stage 0 tiles in flight before the loop.
  if (wave == 0) {
    tdm_load_tile_bf16(A + (size_t)m0 * K, (unsigned)(size_t)&lA[0][0][0], TM);
    tdm_load_tile_bf16(W + (size_t)n0 * K, (unsigned)(size_t)&lB[0][0][0], TN);
  }

  int buf = 0;
  for (int k0 = 0; k0 < K; k0 += TK, buf ^= 1) {
    if (wave == 0) {
      if (k0 + TK < K) {
        tdm_load_tile_bf16(A + (size_t)m0 * K + k0 + TK,
                           (unsigned)(size_t)&lA[buf ^ 1][0][0], TM);
        tdm_load_tile_bf16(W + (size_t)n0 * K + k0 + TK,
                           (unsigned)(size_t)&lB[buf ^ 1][0][0], TN);
        __builtin_amdgcn_s_wait_tensorcnt(2);   // current pair has landed
      } else {
        __builtin_amdgcn_s_wait_tensorcnt(0);
      }
    }
    __syncthreads();

#pragma unroll
    for (int kc = 0; kc < 2; ++kc) {
      // A fragments (16-bit A layout: lanes0-15 K=[0..7,16..23], lanes16-31 +8)
      v16bf af[2];
#pragma unroll
      for (int mt = 0; mt < 2; ++mt) {
        int row = waveM * 32 + mt * 16 + l16;
        af[mt] = cat8(*(const v8bf*)&lA[buf][row][kc * 32 + half * 8],
                      *(const v8bf*)&lA[buf][row][kc * 32 + half * 8 + 16]);
      }
      // B fragments (lane = column n, K=[0..7,16..23], lanes16-31 +8)
#pragma unroll
      for (int nt = 0; nt < 2; ++nt) {
        int row = waveN * 32 + nt * 16 + l16;
        v16bf bfrag = cat8(*(const v8bf*)&lB[buf][row][kc * 32 + half * 8],
                           *(const v8bf*)&lB[buf][row][kc * 32 + half * 8 + 16]);
#pragma unroll
        for (int mt = 0; mt < 2; ++mt)
          acc[mt][nt] = __builtin_amdgcn_wmma_f32_16x16x32_bf16(
              false, af[mt], false, bfrag, (short)0, acc[mt][nt],
              false, false);
      }
    }
    __syncthreads();   // all waves done with buf before TDM overwrites it
  }

  // Store (C layout: VGPR r -> row r + 8*half, lane&15 -> column)
#pragma unroll
  for (int mt = 0; mt < 2; ++mt)
#pragma unroll
    for (int nt = 0; nt < 2; ++nt)
#pragma unroll
      for (int r = 0; r < 8; ++r) {
        int m = m0 + waveM * 32 + mt * 16 + r + half * 8;
        int n = n0 + waveN * 32 + nt * 16 + l16;
        float v = acc[mt][nt][r] + bias[n];
        if (MODE == 1) {                       // [B,H,S,dk]
          int b = m >> 11, s = m & (SEQ - 1);
          int h = n >> 6,  d = n & (DK - 1);
          outb[(((size_t)(b * NH + h) * SEQ + s) << 6) + d] = (__bf16)v;
        } else if (MODE == 2) {                // [B,H,dk,S]  (V transposed)
          int b = m >> 11, s = m & (SEQ - 1);
          int h = n >> 6,  d = n & (DK - 1);
          outb[((size_t)(b * NH + h) * DK + d) * SEQ + s] = (__bf16)v;
        } else {                               // f32 row-major
          outf[(size_t)m * DM + n] = v;
        }
      }
}

// ---------------------------------------------------------------------------
// Flash-attention forward per (b,h): online softmax, all-WMMA.
// Block: 128 threads = 4 waves; each wave owns 16 query rows; block = 64 rows.
// The P-tile LDS scratch is strictly per-wave; CDNA5 keeps a wave's DS ops in
// order, so a wave-local s_wait_dscnt replaces block-wide barriers entirely.
// Q,K: [B,H,S,64] bf16 ; Vt: [B,H,64,S] bf16 ; ctx out: [B,S,1024] bf16.
// ---------------------------------------------------------------------------
__launch_bounds__(128)
__global__ void attn_fwd(const __bf16* __restrict__ Q,
                         const __bf16* __restrict__ Kc,
                         const __bf16* __restrict__ Vt,
                         __bf16* __restrict__ ctx) {
  __shared__ __align__(16) __bf16 pbuf[4][16][72];  // per-wave P tile (padded)

  const int t    = threadIdx.x;
  const int wave = t >> 5, lane = t & 31;
  const int half = lane >> 4, l16 = lane & 15;
  const int h = blockIdx.y, b = blockIdx.z;
  const int bh = b * NH + h;
  const int qbase = blockIdx.x * 64 + wave * 16;

  const __bf16* Qh = Q  + (size_t)bh * SEQ * DK;
  const __bf16* Kh = Kc + (size_t)bh * SEQ * DK;
  const __bf16* Vh = Vt + (size_t)bh * DK * SEQ;

  // Q A-fragments for the whole row block (dk = 64 -> two 16x32 chunks)
  v16bf aq[2];
#pragma unroll
  for (int c = 0; c < 2; ++c) {
    const __bf16* p = Qh + (size_t)(qbase + l16) * DK + 32 * c + half * 8;
    aq[c] = cat8(*(const v8bf*)p, *(const v8bf*)(p + 16));
  }

  v8f o[4] = {};
  float mrow[8], lrow[8];
#pragma unroll
  for (int r = 0; r < 8; ++r) { mrow[r] = -__builtin_inff(); lrow[r] = 0.0f; }

  for (int kb = 0; kb < SEQ; kb += 64) {
    // ---- scores S = Q K^T (16 x 64), B-fragment = K row (table-order dims)
    v8f sc[4] = {};
#pragma unroll
    for (int nt = 0; nt < 4; ++nt) {
#pragma unroll
      for (int c = 0; c < 2; ++c) {
        const __bf16* p =
            Kh + (size_t)(kb + nt * 16 + l16) * DK + 32 * c + half * 8;
        v16bf bk = cat8(*(const v8bf*)p, *(const v8bf*)(p + 16));
        sc[nt] = __builtin_amdgcn_wmma_f32_16x16x32_bf16(
            false, aq[c], false, bk, (short)0, sc[nt], false, false);
      }
    }

    // ---- online softmax (rows live across 16-lane halves)
#pragma unroll
    for (int r = 0; r < 8; ++r) {
      float tm = -__builtin_inff();
#pragma unroll
      for (int nt = 0; nt < 4; ++nt) {
        sc[nt][r] *= 0.125f;               // 1/sqrt(dk)
        tm = fmaxf(tm, sc[nt][r]);
      }
#pragma unroll
      for (int off = 1; off < 16; off <<= 1)
        tm = fmaxf(tm, __shfl_xor(tm, off, 32));
      float mn    = fmaxf(mrow[r], tm);
      float alpha = __expf(mrow[r] - mn);
      mrow[r] = mn;
      lrow[r] *= alpha;
#pragma unroll
      for (int nt = 0; nt < 4; ++nt) o[nt][r] *= alpha;
      float rs = 0.0f;
#pragma unroll
      for (int nt = 0; nt < 4; ++nt) {
        float pv = __expf(sc[nt][r] - mn);
        rs += pv;
        pbuf[wave][r + half * 8][nt * 16 + l16] = (__bf16)pv;  // C-layout
      }
#pragma unroll
      for (int off = 1; off < 16; off <<= 1)
        rs += __shfl_xor(rs, off, 32);
      lrow[r] += rs;
    }

    // Wave-local ordering: this wave's C-layout DS stores must land before
    // its A-layout DS loads (cross-lane through per-wave LDS region).
    asm volatile("s_wait_dscnt 0x0" ::: "memory");

    // ---- O += P V   (P from LDS as A-fragments; Vt gives contiguous keys)
    v16bf pa[2];
#pragma unroll
    for (int c = 0; c < 2; ++c) {
      const __bf16* p = &pbuf[wave][l16][32 * c + half * 8];
      pa[c] = cat8(*(const v8bf*)p, *(const v8bf*)(p + 16));
    }
#pragma unroll
    for (int c = 0; c < 2; ++c)
#pragma unroll
      for (int nt = 0; nt < 4; ++nt) {
        const __bf16* p =
            Vh + (size_t)(nt * 16 + l16) * SEQ + kb + 32 * c + half * 8;
        v16bf bv = cat8(*(const v8bf*)p, *(const v8bf*)(p + 16));
        o[nt] = __builtin_amdgcn_wmma_f32_16x16x32_bf16(
            false, pa[c], false, bv, (short)0, o[nt], false, false);
      }
    // Ensure P reads complete before next iteration overwrites pbuf.
    asm volatile("s_wait_dscnt 0x0" ::: "memory");
  }

  // ---- normalize and write context [B,S,H*dk]
#pragma unroll
  for (int r = 0; r < 8; ++r) {
    float inv = 1.0f / lrow[r];
    int qrow = qbase + r + half * 8;
#pragma unroll
    for (int nt = 0; nt < 4; ++nt) {
      int d = nt * 16 + l16;
      ctx[((size_t)(b * SEQ + qrow)) * DM + h * DK + d] =
          (__bf16)(o[nt][r] * inv);
    }
  }
}

// ---------------------------------------------------------------------------
// Host-side orchestration
// ---------------------------------------------------------------------------
extern "C" void kernel_launch(void* const* d_in, const int* in_sizes, int n_in,
                              void* d_out, int out_size, void* d_ws,
                              size_t ws_size, hipStream_t stream) {
  (void)in_sizes; (void)n_in; (void)out_size; (void)ws_size;

  char* ws = (char*)d_ws;
  size_t off = 0;
  auto carve = [&](size_t bytes) -> char* {
    char* p = ws + off;
    off += (bytes + 255) & ~(size_t)255;
    return p;
  };

  const size_t actBytes = (size_t)MT * DM * sizeof(__bf16);  // 8 MB
  const size_t wBytes   = (size_t)DM * DM * sizeof(__bf16);  // 2 MB

  __bf16* qb  = (__bf16*)carve(actBytes);
  __bf16* kb  = (__bf16*)carve(actBytes);
  __bf16* vb  = (__bf16*)carve(actBytes);
  __bf16* wqb = (__bf16*)carve(wBytes);
  __bf16* wkb = (__bf16*)carve(wBytes);
  __bf16* wvb = (__bf16*)carve(wBytes);
  __bf16* wob = (__bf16*)carve(wBytes);
  __bf16* Qp  = (__bf16*)carve(actBytes);   // [B,H,S,dk]
  __bf16* Kp  = (__bf16*)carve(actBytes);   // [B,H,S,dk]
  __bf16* Vtp = (__bf16*)carve(actBytes);   // [B,H,dk,S]
  __bf16* ctx = (__bf16*)carve(actBytes);   // [B,S,D]

  auto cvt = [&](const void* in, __bf16* out, size_t n) {
    int n4 = (int)(n / 4);
    int blocks = (n4 + 255) / 256;
    if (blocks > 4096) blocks = 4096;
    f32_to_bf16_kernel<<<blocks, 256, 0, stream>>>((const float*)in, out, n4);
  };

  cvt(d_in[0], qb,  (size_t)MT * DM);
  cvt(d_in[1], kb,  (size_t)MT * DM);
  cvt(d_in[2], vb,  (size_t)MT * DM);
  cvt(d_in[3], wqb, (size_t)DM * DM);
  cvt(d_in[5], wkb, (size_t)DM * DM);
  cvt(d_in[7], wvb, (size_t)DM * DM);
  cvt(d_in[9], wob, (size_t)DM * DM);

  dim3 ggrid(DM / TN, MT / TM);   // (8, 64)
  gemm_bf16_wmma<1><<<ggrid, 256, 0, stream>>>(qb, wqb, (const float*)d_in[4],
                                               Qp, nullptr);
  gemm_bf16_wmma<1><<<ggrid, 256, 0, stream>>>(kb, wkb, (const float*)d_in[6],
                                               Kp, nullptr);
  gemm_bf16_wmma<2><<<ggrid, 256, 0, stream>>>(vb, wvb, (const float*)d_in[8],
                                               Vtp, nullptr);

  dim3 agrid(SEQ / 64, NH, BATCH);
  attn_fwd<<<agrid, 128, 0, stream>>>(Qp, Kp, Vtp, ctx);

  gemm_bf16_wmma<3><<<ggrid, 256, 0, stream>>>(ctx, wob, (const float*)d_in[10],
                                               nullptr, (float*)d_out);
}
